// DAN_MoVe_85985245266015
// MI455X (gfx1250) — compile-verified, matrix operand order checked by reading, not run on hardware
//
#include <hip/hip_runtime.h>
#include <hip/hip_bf16.h>

// ---------------------------------------------------------------------------
// Problem constants (from reference): B=2, FRAME=9, SFRAME=15, C=128, VC=512,
// H=W=40 -> HW=1600, WK = 15*1600 = 24000, NQ2 = 9*1600 = 14400, mid = 4.
// ---------------------------------------------------------------------------
#define B_      2
#define FRAME_  9
#define SFRAME_ 15
#define C_      128
#define VC_     512
#define HW_     1600
#define WK_     (SFRAME_ * HW_)   // 24000
#define NQ2_    (FRAME_ * HW_)    // 14400
#define MID_    4

typedef __attribute__((ext_vector_type(16))) __bf16 v16bf;
typedef __attribute__((ext_vector_type(8)))  float  v8f;

__device__ __forceinline__ unsigned short f2bf(float x) {
    unsigned int u = __float_as_uint(x);
    u += 0x7FFFu + ((u >> 16) & 1u);     // round-to-nearest-even
    return (unsigned short)(u >> 16);
}

union TileU {
    v16bf        v;
    uint4        u[2];
    unsigned int w[8];
};

// Load a 16(row) x 32(col) bf16 tile from a row-major [rows][ld] buffer into
// the CDNA5 16-bit A/B matrix VGPR layout:
//   lanes 0-15 : row = lane,    cols {0..7, 16..23}
//   lanes 16-31: row = lane-16, cols {8..15, 24..31}
// Two global_load_b128 per lane. Requires 16B row alignment.
__device__ __forceinline__ v16bf load_tile16x32(const unsigned short* base, int ld) {
    const int lane = threadIdx.x & 31;
    const unsigned short* p = base + (size_t)(lane & 15) * ld + ((lane >> 4) << 3);
    TileU t;
    t.u[0] = *(const uint4*)(p);        // cols +0..7  (or +8..15 for hi lanes)
    t.u[1] = *(const uint4*)(p + 16);   // cols +16..23 (or +24..31)
    return t.v;
}

__device__ __forceinline__ v8f wmma_bf16(v16bf a, v16bf b, v8f c) {
    // D = A(16x32 bf16) * B(32x16 bf16) + C(16x16 f32)
    return __builtin_amdgcn_wmma_f32_16x16x32_bf16(
        /*neg_a=*/false, a, /*neg_b=*/false, b,
        /*c_mod=*/(short)0, c, /*reuse_a=*/false, /*reuse_b=*/false);
}

// ---------------------------------------------------------------------------
// Prep: transpose a [128][1600] f32 matrix into row-major [1600][128] bf16.
// Used for SK (30 mats), QQT (18), QmidT (2), MK (2).
// ---------------------------------------------------------------------------
__global__ void transpose128_bf16(const float* __restrict__ src,
                                  unsigned short* __restrict__ dst,
                                  long nmats, long sstride, long dstride) {
    const long MAT = (long)HW_ * C_;
    long total = nmats * MAT;
    for (long i = (long)blockIdx.x * blockDim.x + threadIdx.x; i < total;
         i += (long)gridDim.x * blockDim.x) {
        long mat = i / MAT;
        long rem = i - mat * MAT;
        int  p   = (int)(rem >> 7);    // 0..1599
        int  c   = (int)(rem & 127);   // 0..127
        dst[mat * dstride + rem] = f2bf(src[mat * sstride + (long)c * HW_ + p]);
    }
}

// Prep: support_v [B][SF][VC][HW] f32  ->  SV [B][VC][SF*HW] bf16
__global__ void permute_sv_bf16(const float* __restrict__ src,
                                unsigned short* __restrict__ dst) {
    long total = (long)B_ * SFRAME_ * VC_ * HW_;
    for (long i = (long)blockIdx.x * blockDim.x + threadIdx.x; i < total;
         i += (long)gridDim.x * blockDim.x) {
        int  p  = (int)(i % HW_);
        long r  = i / HW_;
        int  v  = (int)(r % VC_);
        long r2 = r / VC_;
        int  f  = (int)(r2 % SFRAME_);
        int  b  = (int)(r2 / SFRAME_);
        dst[(((long)b * VC_ + v) * SFRAME_ + f) * HW_ + p] = f2bf(src[i]);
    }
}

// ---------------------------------------------------------------------------
// Fused flash-attention stage.
//   K : [Nk][128] bf16 row-major (keys)          -> WMMA A operand
//   Q : [Nq][128] bf16 row-major (queries^T)     -> WMMA B operand
//   V : [512][Nk] bf16 row-major (values)        -> WMMA A operand
// Workgroup = 8 waves, one 16-query tile; wave w owns value rows [64w,64w+64).
//
// Scheduling (enforced with __builtin_amdgcn_sched_barrier(0) fences so the
// post-RA scheduler cannot sink loads onto their consumers):
//   [K loads] | [S WMMAs, progressive s_wait_loadcnt] | [V loads] |
//   [softmax VALU covers V latency] | [rescale + V WMMAs]
// MODE 0: store bf16 [512][Nq] (new_V).  MODE 1: store final f32 output.
// ---------------------------------------------------------------------------
template <int MODE>
__global__ __launch_bounds__(256)
void attn_kernel(const unsigned short* __restrict__ Kb,
                 const unsigned short* __restrict__ Qb,
                 const unsigned short* __restrict__ Vb,
                 void* __restrict__ outp,
                 int Nk, int Nq, long ksb, long qsb, long vsb) {
    const int b    = blockIdx.y;
    const unsigned short* K = Kb + (size_t)b * ksb;
    const unsigned short* Q = Qb + (size_t)b * qsb;
    const unsigned short* V = Vb + (size_t)b * vsb;

    const int q0   = blockIdx.x * 16;
    const int wave = threadIdx.x >> 5;
    const int lane = threadIdx.x & 31;
    const int v0   = wave * 64;

    // Query B-tiles for the 4 c-chunks, resident for the whole key loop.
    v16bf qB[4];
#pragma unroll
    for (int cc = 0; cc < 4; ++cc)
        qB[cc] = load_tile16x32(Q + (size_t)q0 * C_ + cc * 32, C_);

    v8f acc[4] = {};
    float m = -1e30f, l = 0.f;
    const float sc = 0.08838834764831845f;  // 1/sqrt(128)

    for (int kb = 0; kb < Nk; kb += 32) {
        // ---- phase 1: issue ALL K-tile loads --------------------------------
        v16bf kA0[4], kA1[4];
#pragma unroll
        for (int cc = 0; cc < 4; ++cc) {
            kA0[cc] = load_tile16x32(K + (size_t)kb * C_ + cc * 32, C_);
            kA1[cc] = load_tile16x32(K + (size_t)(kb + 16) * C_ + cc * 32, C_);
        }
        __builtin_amdgcn_sched_barrier(0);   // keep all K loads issued up front

        // ---- phase 2: S WMMAs (in-order VMEM => progressive waits) ----------
        v8f s0 = {}, s1 = {};
#pragma unroll
        for (int cc = 0; cc < 4; ++cc) {
            s0 = wmma_bf16(kA0[cc], qB[cc], s0);
            s1 = wmma_bf16(kA1[cc], qB[cc], s1);
        }
        __builtin_amdgcn_sched_barrier(0);

        // ---- phase 3: issue ALL V-tile loads (independent of S) -------------
        v16bf vA[4];
#pragma unroll
        for (int t = 0; t < 4; ++t)
            vA[t] = load_tile16x32(V + (size_t)(v0 + 16 * t) * Nk + kb, Nk);
        if (kb + 32 < Nk)
            __builtin_prefetch(K + (size_t)(kb + 32) * C_, 0, 1);
        __builtin_amdgcn_sched_barrier(0);   // V loads stay above the softmax

        // ---- phase 4: online softmax (covers V-load latency) ----------------
        float mx = -1e30f;
#pragma unroll
        for (int r = 0; r < 8; ++r) {
            s0[r] *= sc;
            s1[r] *= sc;
            mx = fmaxf(mx, fmaxf(s0[r], s1[r]));
        }
        mx = fmaxf(mx, __shfl_xor(mx, 16, 32));   // lanes l / l+16 cover column
        float mnew  = fmaxf(m, mx);
        float alpha = __expf(m - mnew);
        float sum   = 0.f;

        // exp + pack directly into the 32x16 B-matrix layout: the C-layout
        // rows each lane holds are exactly the (k,n) entries its B operand
        // needs (lo lanes: k {0..7,16..23}, hi lanes: k {8..15,24..31}).
        TileU pB;
#pragma unroll
        for (int j = 0; j < 4; ++j) {
            float a0 = __expf(s0[2 * j]     - mnew);
            float a1 = __expf(s0[2 * j + 1] - mnew);
            float b0 = __expf(s1[2 * j]     - mnew);
            float b1 = __expf(s1[2 * j + 1] - mnew);
            sum += a0 + a1 + b0 + b1;
            pB.w[j]     = (unsigned)f2bf(a0) | ((unsigned)f2bf(a1) << 16);
            pB.w[4 + j] = (unsigned)f2bf(b0) | ((unsigned)f2bf(b1) << 16);
        }
        sum += __shfl_xor(sum, 16, 32);
        l = l * alpha + sum;
        m = mnew;

        // ---- phase 5: rescale accumulators and accumulate V*P ---------------
#pragma unroll
        for (int t = 0; t < 4; ++t) {
#pragma unroll
            for (int r = 0; r < 8; ++r) acc[t][r] *= alpha;
            acc[t] = wmma_bf16(vA[t], pB.v, acc[t]);
        }
    }

    const float inv = 1.f / l;
    const int n       = lane & 15;         // query within tile (C-layout column)
    const int rowbase = (lane >> 4) << 3;  // 0 or 8 (C-layout row half)

    if (MODE == 0) {
        unsigned short* o = (unsigned short*)outp + (size_t)b * VC_ * Nq;
#pragma unroll
        for (int t = 0; t < 4; ++t)
#pragma unroll
            for (int r = 0; r < 8; ++r) {
                int v = v0 + 16 * t + rowbase + r;
                o[(size_t)v * Nq + q0 + n] = f2bf(acc[t][r] * inv);
            }
    } else {
        // out[b][f][v][p] = M[v][f*HW + p]; a 16-query tile never crosses a frame.
        float* o = (float*)outp + (size_t)b * FRAME_ * VC_ * HW_;
        const int f = q0 / HW_;
        const int p = q0 - f * HW_ + n;
#pragma unroll
        for (int t = 0; t < 4; ++t)
#pragma unroll
            for (int r = 0; r < 8; ++r) {
                int v = v0 + 16 * t + rowbase + r;
                o[((size_t)f * VC_ + v) * HW_ + p] = acc[t][r] * inv;
            }
    }
}

// ---------------------------------------------------------------------------
extern "C" void kernel_launch(void* const* d_in, const int* in_sizes, int n_in,
                              void* d_out, int out_size, void* d_ws, size_t ws_size,
                              hipStream_t stream) {
    const float* query_q   = (const float*)d_in[0];  // [2][9][128][1600]
    const float* query_k   = (const float*)d_in[1];  // [2][9][128][1600]
    const float* support_k = (const float*)d_in[2];  // [2][15][128][1600]
    const float* support_v = (const float*)d_in[3];  // [2][15][512][1600]

    char*  ws  = (char*)d_ws;
    size_t off = 0;
    auto take = [&](size_t bytes) -> char* {
        char* p = ws + off;
        off = (off + bytes + 255) & ~(size_t)255;
        return p;
    };

    unsigned short* SK    = (unsigned short*)take((size_t)B_ * WK_  * C_  * 2); // [b][24000][128]
    unsigned short* SV    = (unsigned short*)take((size_t)B_ * VC_  * WK_ * 2); // [b][512][24000]
    unsigned short* QmidT = (unsigned short*)take((size_t)B_ * HW_  * C_  * 2); // [b][1600][128]
    unsigned short* QQT   = (unsigned short*)take((size_t)B_ * NQ2_ * C_  * 2); // [b][14400][128]
    unsigned short* MK    = (unsigned short*)take((size_t)B_ * HW_  * C_  * 2); // [b][1600][128]
    unsigned short* newV  = (unsigned short*)take((size_t)B_ * VC_  * HW_ * 2); // [b][512][1600]

    const long MAT = (long)HW_ * C_;  // 204800 elements per [128][1600] matrix

    // SK: support_k as 30 consecutive [128][1600] mats -> [b][f*HW][128]
    {
        long total = 30L * MAT;
        transpose128_bf16<<<(int)((total + 255) / 256), 256, 0, stream>>>(
            support_k, SK, 30, MAT, MAT);
    }
    // QQT: all 18 query_q frames
    {
        long total = 18L * MAT;
        transpose128_bf16<<<(int)((total + 255) / 256), 256, 0, stream>>>(
            query_q, QQT, 18, MAT, MAT);
    }
    // QmidT: middle frame of query_q per batch
    {
        long total = 2L * MAT;
        transpose128_bf16<<<(int)((total + 255) / 256), 256, 0, stream>>>(
            query_q + (long)MID_ * MAT, QmidT, 2, (long)FRAME_ * MAT, MAT);
    }
    // MK: middle frame of query_k per batch
    {
        long total = 2L * MAT;
        transpose128_bf16<<<(int)((total + 255) / 256), 256, 0, stream>>>(
            query_k + (long)MID_ * MAT, MK, 2, (long)FRAME_ * MAT, MAT);
    }
    // SV: [b][f][v][p] -> [b][v][f*HW]
    {
        long total = (long)B_ * SFRAME_ * VC_ * HW_;
        permute_sv_bf16<<<(int)((total + 255) / 256), 256, 0, stream>>>(support_v, SV);
    }

    // Stage 1: middle-frame queries attend over 24000 support keys -> new_V (bf16)
    {
        dim3 grid(HW_ / 16, B_);
        attn_kernel<0><<<grid, 256, 0, stream>>>(
            SK, QmidT, SV, newV, WK_, HW_,
            (long)WK_ * C_, (long)HW_ * C_, (long)VC_ * WK_);
    }
    // Stage 2: all-frame queries attend over 1600 middle keys -> final f32 output
    {
        dim3 grid(NQ2_ / 16, B_);
        attn_kernel<1><<<grid, 256, 0, stream>>>(
            MK, QQT, newV, d_out, HW_, NQ2_,
            (long)HW_ * C_, (long)NQ2_ * C_, (long)VC_ * HW_);
    }
}
